// BilinearPolicy_9036611191026
// MI455X (gfx1250) — matrix operand order repaired and checked by reading, not compiled, not run on hardware
//
#include <hip/hip_runtime.h>
#include <stdint.h>
#include <stddef.h>

// ---------------------------------------------------------------------------
// MI455X / gfx1250 (CDNA5, wave32) bilinear policy network, round 5.
// bf16x3 split-precision GEMMs on V_WMMA_F32_16X16X32_BF16.
// Double-buffered async global->LDS pipeline (GLOBAL_LOAD_ASYNC_TO_LDS_B128 +
// counted S_WAIT_ASYNCCNT): tile t+1 streams into LDS while tile t is in the
// matrix pipe, so DMA latency hides behind WMMA work.
// ---------------------------------------------------------------------------

typedef __attribute__((ext_vector_type(16))) __bf16          v16bf;
typedef __attribute__((ext_vector_type(8)))  float           v8f;
typedef __attribute__((ext_vector_type(4)))  float           v4f;
typedef __attribute__((ext_vector_type(4)))  unsigned        v4u;
typedef __attribute__((ext_vector_type(4)))  unsigned short  v4us;
typedef __attribute__((ext_vector_type(8)))  unsigned short  v8us;
typedef __attribute__((__vector_size__(4 * sizeof(int)))) int v4i;  // builtin's pointee type

#define LDT 48  // padded LDS k-stride (bf16 elems): 32 data + 16 pad; 96B rows

#ifndef __has_builtin
#define __has_builtin(x) 0
#endif
#if __has_builtin(__builtin_amdgcn_global_load_async_to_lds_b128)
#define HAVE_ASYNC_LDS 1
#else
#define HAVE_ASYNC_LDS 0
#endif

#if HAVE_ASYNC_LDS
__device__ __forceinline__ void async_copy16(const unsigned short* g, unsigned short* l) {
  // per-lane: LDS[l .. l+15] = MEM[g .. g+15], tracked by ASYNCcnt
  __builtin_amdgcn_global_load_async_to_lds_b128((v4i*)g, (v4i*)l, 0, 0);
}
template <int N>
__device__ __forceinline__ void async_wait_le() {
#if __has_builtin(__builtin_amdgcn_s_wait_asynccnt)
  __builtin_amdgcn_s_wait_asynccnt(N);
#else
  asm volatile("s_wait_asynccnt %0" :: "i"(N) : "memory");
#endif
}
#endif

__device__ __forceinline__ unsigned short f32_to_bf16(float x) {
  unsigned u = __float_as_uint(x);
  u += 0x7FFFu + ((u >> 16) & 1u);   // round-to-nearest-even
  return (unsigned short)(u >> 16);
}
__device__ __forceinline__ float bf16_to_f32(unsigned short h) {
  return __uint_as_float(((unsigned)h) << 16);
}

struct B32x8 { v4u lo, hi; };  // 32 bytes == one v16bf fragment

// A fragment (16x32 bf16): lanes 0-15 = rows, K={0..7,16..23}; lanes 16-31 same
// rows, K={8..15,24..31}.  Two contiguous 16B LDS chunks per lane.
__device__ __forceinline__ v16bf frag_a(const unsigned short* tile, int r, int h) {
  const unsigned short* p = tile + r * LDT + h * 8;
  B32x8 s;
  s.lo = *(const v4u*)(p);
  s.hi = *(const v4u*)(p + 16);
  return __builtin_bit_cast(v16bf, s);
}
// B fragment (32x16 bf16): lane n = column N; lanes 0-15 K={0..7,16..23}, 16-31 K={8..15,24..31}.
// B staged transposed (N-major) -> two 16B LDS chunks per lane.
__device__ __forceinline__ v16bf frag_b(const unsigned short* tile, int n, int h) {
  const unsigned short* p = tile + n * LDT + h * 8;
  B32x8 s;
  s.lo = *(const v4u*)(p);
  s.hi = *(const v4u*)(p + 16);
  return __builtin_bit_cast(v16bf, s);
}

#define WMMA_BF16(A, B, C) \
  __builtin_amdgcn_wmma_f32_16x16x32_bf16(false, (A), false, (B), (short)0, (C), false, false)

// ---------------------------------------------------------------------------
// FC layer on bf16 hi/lo planes:
//   Y = act(X @ W + b),  X as (Xh,Xl)[M,K],  W as transposed planes (Wth,Wtl)[N,K]
// Output written as bf16 hi/lo planes (converted once, here).
// Workgroup tile 128x64, 8 waves 4(M)x2(N), wave tile 32x32.
// Double-buffered async pipeline; 6 async b128 per thread per tile.
// ---------------------------------------------------------------------------
__global__ __launch_bounds__(256) void fc_bf16_kernel(
    const unsigned short* __restrict__ Xh, const unsigned short* __restrict__ Xl,
    const unsigned short* __restrict__ Wth, const unsigned short* __restrict__ Wtl,
    const float* __restrict__ bias,
    unsigned short* __restrict__ Yh, unsigned short* __restrict__ Yl,
    int K, int N, int do_relu)
{
  __shared__ __align__(16) unsigned short sAh[2][128 * LDT];
  __shared__ __align__(16) unsigned short sAl[2][128 * LDT];
  __shared__ __align__(16) unsigned short sBh[2][64 * LDT];
  __shared__ __align__(16) unsigned short sBl[2][64 * LDT];

  const int tid  = threadIdx.x;
  const int bm0  = blockIdx.y * 128;
  const int bn0  = blockIdx.x * 64;
  const int lane = tid & 31;
  const int wv   = tid >> 5;
  const int wm   = wv >> 1, wn = wv & 1;
  const int r    = lane & 15, h = lane >> 4;

  v8f acc[2][2] = {};
  const int nk = K >> 5;

#if HAVE_ASYNC_LDS
  auto stage = [&](int k0, int b) {
#pragma unroll
    for (int i = 0; i < 2; ++i) {
      int e = tid + i * 256;
      int row = e >> 2, cq = e & 3;
      size_t g = (size_t)(bm0 + row) * K + k0 + cq * 8;
      int off = row * LDT + cq * 8;
      async_copy16(Xh + g, &sAh[b][off]);
      async_copy16(Xl + g, &sAl[b][off]);
    }
    {
      int row = tid >> 2, cq = tid & 3;
      size_t g = (size_t)(bn0 + row) * K + k0 + cq * 8;
      int off = row * LDT + cq * 8;
      async_copy16(Wth + g, &sBh[b][off]);
      async_copy16(Wtl + g, &sBl[b][off]);
    }
  };
  stage(0, 0);   // pipeline prologue
#endif

  for (int t = 0; t < nk; ++t) {
    int cur = t & 1;
    const int k0 = t << 5;
#if HAVE_ASYNC_LDS
    if (t + 1 < nk) { stage((t + 1) << 5, cur ^ 1); async_wait_le<6>(); }
    else            { async_wait_le<0>(); }
#else
    cur = 0;
    // batched register staging: cluster all loads, then all LDS stores
    v8us rAh[2], rAl[2], rBh, rBl;
    int offA[2], offB;
#pragma unroll
    for (int i = 0; i < 2; ++i) {
      int e = tid + i * 256;
      int row = e >> 2, cq = e & 3;
      size_t g = (size_t)(bm0 + row) * K + k0 + cq * 8;
      offA[i] = row * LDT + cq * 8;
      rAh[i] = *(const v8us*)(Xh + g);
      rAl[i] = *(const v8us*)(Xl + g);
    }
    {
      int row = tid >> 2, cq = tid & 3;
      size_t g = (size_t)(bn0 + row) * K + k0 + cq * 8;
      offB = row * LDT + cq * 8;
      rBh = *(const v8us*)(Wth + g);
      rBl = *(const v8us*)(Wtl + g);
    }
#pragma unroll
    for (int i = 0; i < 2; ++i) {
      *(v8us*)(&sAh[0][offA[i]]) = rAh[i];
      *(v8us*)(&sAl[0][offA[i]]) = rAl[i];
    }
    *(v8us*)(&sBh[0][offB]) = rBh;
    *(v8us*)(&sBl[0][offB]) = rBl;
#endif
    __syncthreads();

    v16bf ah[2], al[2], bh[2], bl[2];
#pragma unroll
    for (int mi = 0; mi < 2; ++mi) {
      ah[mi] = frag_a(&sAh[cur][(wm * 32 + mi * 16) * LDT], r, h);
      al[mi] = frag_a(&sAl[cur][(wm * 32 + mi * 16) * LDT], r, h);
    }
#pragma unroll
    for (int ni = 0; ni < 2; ++ni) {
      bh[ni] = frag_b(&sBh[cur][(wn * 32 + ni * 16) * LDT], r, h);
      bl[ni] = frag_b(&sBl[cur][(wn * 32 + ni * 16) * LDT], r, h);
    }
#pragma unroll
    for (int mi = 0; mi < 2; ++mi)
#pragma unroll
      for (int ni = 0; ni < 2; ++ni) {
        acc[mi][ni] = WMMA_BF16(ah[mi], bh[ni], acc[mi][ni]);   // hi*hi
        acc[mi][ni] = WMMA_BF16(ah[mi], bl[ni], acc[mi][ni]);   // hi*lo
        acc[mi][ni] = WMMA_BF16(al[mi], bh[ni], acc[mi][ni]);   // lo*hi
      }
    __syncthreads();   // protects buffer reuse by iteration t+1's stage
  }

  // ---- epilogue: +bias, ReLU, single fp32->bf16(hi,lo) split per element ----
#pragma unroll
  for (int mi = 0; mi < 2; ++mi)
#pragma unroll
    for (int ni = 0; ni < 2; ++ni) {
      int col = bn0 + wn * 32 + ni * 16 + r;
      float bv = bias[col];
#pragma unroll
      for (int v = 0; v < 8; ++v) {
        int row = bm0 + wm * 32 + mi * 16 + h * 8 + v;
        float val = acc[mi][ni][v] + bv;
        if (do_relu) val = fmaxf(val, 0.0f);
        unsigned short hb = f32_to_bf16(val);
        size_t g = (size_t)row * N + col;
        Yh[g] = hb;
        Yl[g] = f32_to_bf16(val - bf16_to_f32(hb));
      }
    }
}

// ---------------------------------------------------------------------------
// Fused final layers of BOTH trunks + bilinear diagonal:
//   pred[b,a] = sum_f (Hob@Wo[:,a*512+f]+bo) * (Hdl@Wd[:,f*32+a]+bd)
// Weight planes pre-transposed so every output column c has a contiguous
// [c][K] bf16 row:  Wot[n][k]  and  Wdpt[(a*512+f)][k].
// One workgroup = 128 batch rows x 1 action, f chunked by 64.
// Double-buffered async pipeline; 12 async b128 per thread per tile.
// ---------------------------------------------------------------------------
__global__ __launch_bounds__(256) void final_bilinear_kernel(
    const unsigned short* __restrict__ Hobh, const unsigned short* __restrict__ Hobl,
    const unsigned short* __restrict__ Hdlh, const unsigned short* __restrict__ Hdll,
    const unsigned short* __restrict__ Woth, const unsigned short* __restrict__ Wotl,
    const float* __restrict__ bo,
    const unsigned short* __restrict__ Wdpth, const unsigned short* __restrict__ Wdptl,
    const float* __restrict__ bdp,
    float* __restrict__ pred)
{
  const int K = 2048, FE = 512, ACT = 32;
  const int NK = K >> 5;   // 64 k-tiles per f-chunk

  __shared__ __align__(16) unsigned short sAoh[2][128 * LDT];
  __shared__ __align__(16) unsigned short sAol[2][128 * LDT];
  __shared__ __align__(16) unsigned short sAdh[2][128 * LDT];
  __shared__ __align__(16) unsigned short sAdl[2][128 * LDT];
  __shared__ __align__(16) unsigned short sBoh[2][64 * LDT];
  __shared__ __align__(16) unsigned short sBol[2][64 * LDT];
  __shared__ __align__(16) unsigned short sBdh[2][64 * LDT];
  __shared__ __align__(16) unsigned short sBdl[2][64 * LDT];

  const int a    = blockIdx.x;          // action 0..31
  const int bm0  = blockIdx.y * 128;    // batch tile
  const int tid  = threadIdx.x;
  const int lane = tid & 31;
  const int wv   = tid >> 5;
  const int wm   = wv >> 1, wn = wv & 1;
  const int r    = lane & 15, h = lane >> 4;

  float rowacc[2][8] = {};

#if HAVE_ASYNC_LDS
  auto stage = [&](int k0, int b, size_t colbase) {
#pragma unroll
    for (int i = 0; i < 2; ++i) {
      int e = tid + i * 256;
      int row = e >> 2, cq = e & 3;
      size_t g = (size_t)(bm0 + row) * K + k0 + cq * 8;
      int off = row * LDT + cq * 8;
      async_copy16(Hobh + g, &sAoh[b][off]);
      async_copy16(Hobl + g, &sAol[b][off]);
      async_copy16(Hdlh + g, &sAdh[b][off]);
      async_copy16(Hdll + g, &sAdl[b][off]);
    }
    {
      int row = tid >> 2, cq = tid & 3;
      size_t g = (colbase + row) * K + k0 + cq * 8;
      int off = row * LDT + cq * 8;
      async_copy16(Woth + g,  &sBoh[b][off]);
      async_copy16(Wotl + g,  &sBol[b][off]);
      async_copy16(Wdpth + g, &sBdh[b][off]);
      async_copy16(Wdptl + g, &sBdl[b][off]);
    }
  };
#endif

  for (int f0 = 0; f0 < FE; f0 += 64) {
    v8f ao[2][2] = {};
    v8f ad[2][2] = {};
    const size_t colbase = (size_t)a * FE + f0;   // row base into [16384][K] planes

#if HAVE_ASYNC_LDS
    stage(0, 0, colbase);   // chunk prologue
#endif

    for (int t = 0; t < NK; ++t) {
      int cur = t & 1;
      const int k0 = t << 5;
#if HAVE_ASYNC_LDS
      if (t + 1 < NK) { stage((t + 1) << 5, cur ^ 1, colbase); async_wait_le<12>(); }
      else            { async_wait_le<0>(); }
#else
      cur = 0;
      v8us rA[2][4], rB[4];
      int offA[2], offB;
#pragma unroll
      for (int i = 0; i < 2; ++i) {
        int e = tid + i * 256;
        int row = e >> 2, cq = e & 3;
        size_t g = (size_t)(bm0 + row) * K + k0 + cq * 8;
        offA[i] = row * LDT + cq * 8;
        rA[i][0] = *(const v8us*)(Hobh + g);
        rA[i][1] = *(const v8us*)(Hobl + g);
        rA[i][2] = *(const v8us*)(Hdlh + g);
        rA[i][3] = *(const v8us*)(Hdll + g);
      }
      {
        int row = tid >> 2, cq = tid & 3;
        size_t g = (colbase + row) * K + k0 + cq * 8;
        offB = row * LDT + cq * 8;
        rB[0] = *(const v8us*)(Woth + g);
        rB[1] = *(const v8us*)(Wotl + g);
        rB[2] = *(const v8us*)(Wdpth + g);
        rB[3] = *(const v8us*)(Wdptl + g);
      }
#pragma unroll
      for (int i = 0; i < 2; ++i) {
        *(v8us*)(&sAoh[0][offA[i]]) = rA[i][0];
        *(v8us*)(&sAol[0][offA[i]]) = rA[i][1];
        *(v8us*)(&sAdh[0][offA[i]]) = rA[i][2];
        *(v8us*)(&sAdl[0][offA[i]]) = rA[i][3];
      }
      *(v8us*)(&sBoh[0][offB]) = rB[0];
      *(v8us*)(&sBol[0][offB]) = rB[1];
      *(v8us*)(&sBdh[0][offB]) = rB[2];
      *(v8us*)(&sBdl[0][offB]) = rB[3];
#endif
      __syncthreads();

      {  // ob trunk
        v16bf ah[2], al[2], bh[2], bl[2];
#pragma unroll
        for (int mi = 0; mi < 2; ++mi) {
          ah[mi] = frag_a(&sAoh[cur][(wm * 32 + mi * 16) * LDT], r, h);
          al[mi] = frag_a(&sAol[cur][(wm * 32 + mi * 16) * LDT], r, h);
        }
#pragma unroll
        for (int ni = 0; ni < 2; ++ni) {
          bh[ni] = frag_b(&sBoh[cur][(wn * 32 + ni * 16) * LDT], r, h);
          bl[ni] = frag_b(&sBol[cur][(wn * 32 + ni * 16) * LDT], r, h);
        }
#pragma unroll
        for (int mi = 0; mi < 2; ++mi)
#pragma unroll
          for (int ni = 0; ni < 2; ++ni) {
            ao[mi][ni] = WMMA_BF16(ah[mi], bh[ni], ao[mi][ni]);
            ao[mi][ni] = WMMA_BF16(ah[mi], bl[ni], ao[mi][ni]);
            ao[mi][ni] = WMMA_BF16(al[mi], bh[ni], ao[mi][ni]);
          }
      }
      {  // dl trunk
        v16bf ah[2], al[2], bh[2], bl[2];
#pragma unroll
        for (int mi = 0; mi < 2; ++mi) {
          ah[mi] = frag_a(&sAdh[cur][(wm * 32 + mi * 16) * LDT], r, h);
          al[mi] = frag_a(&sAdl[cur][(wm * 32 + mi * 16) * LDT], r, h);
        }
#pragma unroll
        for (int ni = 0; ni < 2; ++ni) {
          bh[ni] = frag_b(&sBdh[cur][(wn * 32 + ni * 16) * LDT], r, h);
          bl[ni] = frag_b(&sBdl[cur][(wn * 32 + ni * 16) * LDT], r, h);
        }
#pragma unroll
        for (int mi = 0; mi < 2; ++mi)
#pragma unroll
          for (int ni = 0; ni < 2; ++ni) {
            ad[mi][ni] = WMMA_BF16(ah[mi], bh[ni], ad[mi][ni]);
            ad[mi][ni] = WMMA_BF16(ah[mi], bl[ni], ad[mi][ni]);
            ad[mi][ni] = WMMA_BF16(al[mi], bh[ni], ad[mi][ni]);
          }
      }
      __syncthreads();   // protects buffer reuse by iteration t+1's stage
    }

    // chunk epilogue: +bias, elementwise product, per-lane partial
#pragma unroll
    for (int ni = 0; ni < 2; ++ni) {
      int f = f0 + wn * 32 + ni * 16 + r;
      float bov = bo[a * FE + f];
      float bdv = bdp[a * FE + f];
#pragma unroll
      for (int mi = 0; mi < 2; ++mi)
#pragma unroll
        for (int v = 0; v < 8; ++v)
          rowacc[mi][v] += (ao[mi][ni][v] + bov) * (ad[mi][ni][v] + bdv);
    }
  }

  // reduce over the 16 lanes sharing a row (wave32 butterfly, width 16)
#pragma unroll
  for (int mi = 0; mi < 2; ++mi)
#pragma unroll
    for (int v = 0; v < 8; ++v) {
      float s = rowacc[mi][v];
      s += __shfl_xor(s, 1, 16);
      s += __shfl_xor(s, 2, 16);
      s += __shfl_xor(s, 4, 16);
      s += __shfl_xor(s, 8, 16);
      if (r == 0) {
        int row = bm0 + wm * 32 + mi * 16 + h * 8 + v;
        atomicAdd(&pred[row * ACT + a], s);
      }
    }
}

// ---------------------------------------------------------------------------
// One-time conversion kernels (off the critical loop).
// ---------------------------------------------------------------------------

// fp32[K][N] -> transposed bf16 hi/lo planes [N][K] via LDS 32x32 tile.
// blockIdx.z selects a batch slice (stride K*N on both sides).
__global__ __launch_bounds__(256) void conv_transpose_kernel(
    const float* __restrict__ in, unsigned short* __restrict__ oh,
    unsigned short* __restrict__ ol, int K, int N)
{
  __shared__ float tile[32][33];
  const size_t bofs = (size_t)blockIdx.z * K * N;
  in += bofs; oh += bofs; ol += bofs;

  const int n0 = blockIdx.x * 32;
  const int k0 = blockIdx.y * 32;
  const int tid = threadIdx.x;

  {
    int krow = tid >> 3, nq = tid & 7;
    v4f v = *(const v4f*)(in + (size_t)(k0 + krow) * N + n0 + nq * 4);
#pragma unroll
    for (int j = 0; j < 4; ++j) tile[nq * 4 + j][krow] = v[j];
  }
  __syncthreads();
  {
    int n_loc = tid >> 3, kq = tid & 7;
    v4us hq, lq;
#pragma unroll
    for (int j = 0; j < 4; ++j) {
      float x = tile[n_loc][kq * 4 + j];
      unsigned short hb = f32_to_bf16(x);
      hq[j] = hb;
      lq[j] = f32_to_bf16(x - bf16_to_f32(hb));
    }
    size_t g = (size_t)(n0 + n_loc) * K + k0 + kq * 4;
    *(v4us*)(oh + g) = hq;
    *(v4us*)(ol + g) = lq;
  }
}

// elementwise fp32 -> bf16 hi/lo planes (for obs / deltas)
__global__ __launch_bounds__(256) void conv_pair_kernel(
    const float* __restrict__ in, unsigned short* __restrict__ oh,
    unsigned short* __restrict__ ol, int n4)
{
  int i = blockIdx.x * 256 + threadIdx.x;
  if (i >= n4) return;
  v4f v = *(const v4f*)(in + (size_t)i * 4);
  v4us hq, lq;
#pragma unroll
  for (int j = 0; j < 4; ++j) {
    unsigned short hb = f32_to_bf16(v[j]);
    hq[j] = hb;
    lq[j] = f32_to_bf16(v[j] - bf16_to_f32(hb));
  }
  *(v4us*)(oh + (size_t)i * 4) = hq;
  *(v4us*)(ol + (size_t)i * 4) = lq;
}

// Repack dlt_W3 (fp32): Wdp[a][k][f] = Wd[k][f*32+a]; LDS transpose, coalesced.
__global__ __launch_bounds__(256) void pack_w3_kernel(
    const float* __restrict__ Wd, float* __restrict__ Wdp)
{
  __shared__ float tile[32][33];
  const int k  = blockIdx.y;
  const int f0 = blockIdx.x * 32;
  const int tid = threadIdx.x;

  v4f v = *(const v4f*)(Wd + (size_t)k * 16384 + (size_t)f0 * 32 + tid * 4);
  int f_loc = tid >> 3;
  int a0    = (tid & 7) * 4;
#pragma unroll
  for (int j = 0; j < 4; ++j) tile[a0 + j][f_loc] = v[j];
  __syncthreads();

  int a  = tid >> 3;
  int fq = tid & 7;
  v4f w;
#pragma unroll
  for (int j = 0; j < 4; ++j) w[j] = tile[a][fq * 4 + j];
  *(v4f*)(Wdp + ((size_t)a * 2048 + k) * 512 + f0 + fq * 4) = w;
}

__global__ __launch_bounds__(256) void pack_b3_kernel(
    const float* __restrict__ bd, float* __restrict__ bdp)
{
  int o = blockIdx.x * 256 + threadIdx.x;   // 16384 total
  int a = o >> 9, f = o & 511;
  bdp[o] = bd[f * 32 + a];
}

__global__ __launch_bounds__(256) void zero_kernel(float* __restrict__ p, int n)
{
  int i = blockIdx.x * 256 + threadIdx.x;
  if (i < n) p[i] = 0.0f;
}

// ---------------------------------------------------------------------------
// Launcher.  setup_inputs() dict order:
//  0 obs  1 deltas
//  2 obs_W0  3 obs_b0   4 dlt_W0  5 dlt_b0
//  6 obs_W1  7 obs_b1   8 dlt_W1  9 dlt_b1
// 10 obs_W2 11 obs_b2  12 dlt_W2 13 dlt_b2
// 14 obs_W3 15 obs_b3  16 dlt_W3 17 dlt_b3
// ---------------------------------------------------------------------------
extern "C" void kernel_launch(void* const* d_in, const int* in_sizes, int n_in,
                              void* d_out, int out_size, void* d_ws, size_t ws_size,
                              hipStream_t stream)
{
  (void)in_sizes; (void)n_in; (void)out_size; (void)ws_size;

  const float* obs    = (const float*)d_in[0];
  const float* deltas = (const float*)d_in[1];
  const float* oW[4]  = {(const float*)d_in[2],  (const float*)d_in[6],
                         (const float*)d_in[10], (const float*)d_in[14]};
  const float* ob_[4] = {(const float*)d_in[3],  (const float*)d_in[7],
                         (const float*)d_in[11], (const float*)d_in[15]};
  const float* dW[4]  = {(const float*)d_in[4],  (const float*)d_in[8],
                         (const float*)d_in[12], (const float*)d_in[16]};
  const float* db_[4] = {(const float*)d_in[5],  (const float*)d_in[9],
                         (const float*)d_in[13], (const float*)d_in[17]};
  float* pred = (float*)d_out;

  // ---- workspace bump allocator ----
  char* p = (char*)d_ws;
  auto alloc_b = [&](size_t bytes) { char* r = p; p += (bytes + 255) & ~(size_t)255; return r; };

  const size_t HE = (size_t)4096 * 2048;   // activation elems per plane

  // activation region first (also reused as fp32 Wdp temp before GEMMs run)
  unsigned short* bufH[4]; unsigned short* bufL[4];
  char* act_base = alloc_b(4 * HE * 4);    // 4 buffers x (hi+lo) x 2B = 128 MB
  for (int i = 0; i < 4; ++i) {
    bufH[i] = (unsigned short*)(act_base + (size_t)i * HE * 4);
    bufL[i] = bufH[i] + HE;
  }
  float* wdp_f32 = (float*)act_base;       // 2048*16384 fp32 = 128 MB (temp)

  const size_t IE = (size_t)4096 * 256;
  unsigned short* obsH = (unsigned short*)alloc_b(IE * 2);
  unsigned short* obsL = (unsigned short*)alloc_b(IE * 2);
  unsigned short* dltH = (unsigned short*)alloc_b(IE * 2);
  unsigned short* dltL = (unsigned short*)alloc_b(IE * 2);

  // transposed bf16 weight planes [N][K]
  const size_t WSZ[3] = {(size_t)256 * 2048, (size_t)2048 * 2048, (size_t)2048 * 2048};
  unsigned short *oWtH[3], *oWtL[3], *dWtH[3], *dWtL[3];
  for (int i = 0; i < 3; ++i) {
    oWtH[i] = (unsigned short*)alloc_b(WSZ[i] * 2);
    oWtL[i] = (unsigned short*)alloc_b(WSZ[i] * 2);
    dWtH[i] = (unsigned short*)alloc_b(WSZ[i] * 2);
    dWtL[i] = (unsigned short*)alloc_b(WSZ[i] * 2);
  }
  const size_t W3E = (size_t)2048 * 16384;
  unsigned short* WotH  = (unsigned short*)alloc_b(W3E * 2);
  unsigned short* WotL  = (unsigned short*)alloc_b(W3E * 2);
  unsigned short* WdptH = (unsigned short*)alloc_b(W3E * 2);
  unsigned short* WdptL = (unsigned short*)alloc_b(W3E * 2);
  float* bdp = (float*)alloc_b(16384 * 4);

  // ---- 1) one-time packs / conversions ----
  pack_w3_kernel<<<dim3(16, 2048), 256, 0, stream>>>(dW[3], wdp_f32);
  conv_transpose_kernel<<<dim3(512 / 32, 2048 / 32, 32), 256, 0, stream>>>(
      wdp_f32, WdptH, WdptL, 2048, 512);
  pack_b3_kernel<<<64, 256, 0, stream>>>(db_[3], bdp);

  conv_pair_kernel<<<(int)(IE / 4 + 255) / 256, 256, 0, stream>>>(obs, obsH, obsL, (int)(IE / 4));
  conv_pair_kernel<<<(int)(IE / 4 + 255) / 256, 256, 0, stream>>>(deltas, dltH, dltL, (int)(IE / 4));

  const int KD[3] = {256, 2048, 2048};
  for (int i = 0; i < 3; ++i) {
    dim3 g(2048 / 32, KD[i] / 32, 1);
    conv_transpose_kernel<<<g, 256, 0, stream>>>(oW[i], oWtH[i], oWtL[i], KD[i], 2048);
    conv_transpose_kernel<<<g, 256, 0, stream>>>(dW[i], dWtH[i], dWtL[i], KD[i], 2048);
  }
  conv_transpose_kernel<<<dim3(16384 / 32, 2048 / 32, 1), 256, 0, stream>>>(
      oW[3], WotH, WotL, 2048, 16384);

  // ---- 2) zero output accumulator ----
  zero_kernel<<<512, 256, 0, stream>>>(pred, 4096 * 32);

  // ---- 3) trunks (GEMMs overwrite the act region; temp no longer needed) ----
  const dim3 gg(2048 / 64, 4096 / 128);   // (32, 32)

  fc_bf16_kernel<<<gg, 256, 0, stream>>>(obsH, obsL, oWtH[0], oWtL[0], ob_[0],
                                         bufH[0], bufL[0], 256, 2048, 1);
  fc_bf16_kernel<<<gg, 256, 0, stream>>>(bufH[0], bufL[0], oWtH[1], oWtL[1], ob_[1],
                                         bufH[1], bufL[1], 2048, 2048, 1);
  fc_bf16_kernel<<<gg, 256, 0, stream>>>(bufH[1], bufL[1], oWtH[2], oWtL[2], ob_[2],
                                         bufH[2], bufL[2], 2048, 2048, 1);

  fc_bf16_kernel<<<gg, 256, 0, stream>>>(dltH, dltL, dWtH[0], dWtL[0], db_[0],
                                         bufH[0], bufL[0], 256, 2048, 1);
  fc_bf16_kernel<<<gg, 256, 0, stream>>>(bufH[0], bufL[0], dWtH[1], dWtL[1], db_[1],
                                         bufH[1], bufL[1], 2048, 2048, 1);
  fc_bf16_kernel<<<gg, 256, 0, stream>>>(bufH[1], bufL[1], dWtH[2], dWtL[2], db_[2],
                                         bufH[3], bufL[3], 2048, 2048, 1);

  // ---- 4) fused final layers + bilinear diagonal -> pred[4096,32] ----
  final_bilinear_kernel<<<dim3(32, 32), 256, 0, stream>>>(
      bufH[2], bufL[2], bufH[3], bufL[3],
      WotH, WotL, ob_[3], WdptH, WdptL, bdp, pred);
}
